// ParallelEncoderBlock_22119081574750
// MI455X (gfx1250) — matrix operand order repaired and checked by reading, not compiled, not run on hardware
//
#include <hip/hip_runtime.h>
#include <hip/hip_bf16.h>
#include <stdint.h>

typedef __bf16 bf16_t;
typedef unsigned int u32;
typedef unsigned long long u64;
typedef __attribute__((ext_vector_type(16))) __bf16 v16bf;
typedef __attribute__((ext_vector_type(8)))  float   v8f;
typedef __attribute__((ext_vector_type(4)))  u32     u32x4;
typedef __attribute__((ext_vector_type(8)))  int     i32x8;
typedef __attribute__((ext_vector_type(4)))  int     i32x4;

// AS-qualified vector pointers for the async-to-LDS builtins
typedef __attribute__((address_space(1))) i32x4 i32x4_g;   // global (AS1)
typedef __attribute__((address_space(3))) i32x4 i32x4_l;   // LDS    (AS3)

union V16BF { v16bf v; uint4 u[2]; };

constexpr int kB  = 2;
constexpr int kS  = 2048;
constexpr int kD  = 1024;
constexpr int kH  = 16;
constexpr int kHD = 64;
constexpr int kFF = 4096;
constexpr int kNSUM = 3 * kD + 2 * kFF;   // 11264
constexpr int kM  = kB * kS;              // 4096

__device__ __forceinline__ float  bf2f(bf16_t x) { return (float)x; }
__device__ __forceinline__ bf16_t f2bf(float  x) { return (bf16_t)x; }

// ---------------------------------------------------------------------------
// CDNA5 data-mover feature detection (compile-safe fallbacks)
// ---------------------------------------------------------------------------
#if __has_builtin(__builtin_amdgcn_tensor_load_to_lds)
#define TDM_VARIANT 1
#elif __has_builtin(__builtin_amdgcn_tensor_load_to_lds_d2)
#define TDM_VARIANT 2
#else
#define TDM_VARIANT 0
#endif
#define HAVE_TDM (TDM_VARIANT != 0)

#if __has_builtin(__builtin_amdgcn_global_load_async_to_lds_b128) && __has_builtin(__builtin_amdgcn_s_wait_asynccnt)
#define HAVE_ASYNC 1
#else
#define HAVE_ASYNC 0
#endif

__device__ __forceinline__ u32 lds_off(const void* p) {
  // generic LDS pointer: addr[31:0] is the LDS byte offset (ISA aperture rules)
  return (u32)(uintptr_t)p;
}

#if HAVE_TDM
template <int N>
__device__ __forceinline__ void tdm_wait() {
#if __has_builtin(__builtin_amdgcn_s_wait_tensorcnt)
  __builtin_amdgcn_s_wait_tensorcnt(N);
#else
  asm volatile("s_wait_tensorcnt %0" :: "i"(N) : "memory");
#endif
}

// 2D tile DMA: tile_h rows x tile_w elements (bf16), row stride = stride elems
__device__ __forceinline__ void tdm_load_2d(u32 lds_byte, const void* gaddr,
                                            u32 tile_w, u32 tile_h, u32 stride) {
  u64 ga = (u64)(uintptr_t)gaddr;
  u32x4 g0;
  g0[0] = 1u;                                          // count=1 (user descriptor)
  g0[1] = lds_byte;                                    // lds_addr
  g0[2] = (u32)ga;                                     // global_addr[31:0]
  g0[3] = (u32)((ga >> 32) & 0x1FFFFFFu) | (2u << 30); // global_addr[56:32] | type=2
  u32 tw = tile_w, th = tile_h;
  i32x8 g1;
  g1[0] = (int)(1u << 16);                             // data_size=1 (2B), mask=0
  g1[1] = (int)((tw & 0xFFFFu) << 16);                 // tensor_dim0[15:0]
  g1[2] = (int)((tw >> 16) | ((th & 0xFFFFu) << 16));  // tensor_dim0 hi | tensor_dim1 lo
  g1[3] = (int)((th >> 16) | (tw << 16));              // tensor_dim1 hi | tile_dim0
  g1[4] = (int)th;                                     // tile_dim1 | tile_dim2=0
  g1[5] = (int)stride;                                 // tensor_dim0_stride[31:0]
  g1[6] = 0;                                           // stride hi | dim1_stride lo
  g1[7] = 0;
#if TDM_VARIANT == 1
  i32x4 z = {0, 0, 0, 0};
#if __clang_major__ >= 23
  i32x8 z8 = {0, 0, 0, 0, 0, 0, 0, 0};
  __builtin_amdgcn_tensor_load_to_lds(g0, g1, z, z, z8, 0);
#else
  __builtin_amdgcn_tensor_load_to_lds(g0, g1, z, z, 0);
#endif
#else
  __builtin_amdgcn_tensor_load_to_lds_d2(g0, g1, 0);
#endif
}
#endif  // HAVE_TDM

#if HAVE_ASYNC
__device__ __forceinline__ void async_copy_b128(void* lds, const void* g) {
  __builtin_amdgcn_global_load_async_to_lds_b128(
      (i32x4_g*)(uintptr_t)g, (i32x4_l*)(u32)(uintptr_t)lds, 0, 0);
}
#endif

// stage a ROWSx(U4*8) bf16 tile into LDS (contiguous) from row-major global
template <int ROWS, int U4>
__device__ __forceinline__ void stage_tile(bf16_t* dst, const bf16_t* g, int ld,
                                           int tid, int wave) {
#if HAVE_TDM
  if (wave == 0) tdm_load_2d(lds_off(dst), g, U4 * 8, ROWS, (u32)ld);
#elif HAVE_ASYNC
  for (int idx = tid; idx < ROWS * U4; idx += 128) {
    int r = idx / U4, c = idx % U4;
    async_copy_b128(&dst[idx * 8], g + (size_t)r * ld + c * 8);
  }
#else
  for (int idx = tid; idx < ROWS * U4; idx += 128) {
    int r = idx / U4, c = idx % U4;
    *(uint4*)&dst[idx * 8] = *(const uint4*)(g + (size_t)r * ld + c * 8);
  }
#endif
}

// wait for staged tiles: N_* = outstanding ops allowed when more tiles in flight
template <int N_TDM, int N_ASYNC>
__device__ __forceinline__ void stage_wait(bool more, int wave) {
#if HAVE_TDM
  if (wave == 0) {
    if (more) tdm_wait<N_TDM>();
    else      tdm_wait<0>();
  }
#elif HAVE_ASYNC
  if (more) __builtin_amdgcn_s_wait_asynccnt(N_ASYNC);
  else      __builtin_amdgcn_s_wait_asynccnt(0);
#endif
  __syncthreads();
}

// ---------------------------------------------------------------------------
// fp32 -> bf16 transposed conversion: dst[n*K + k] = src[k*N + n]
// ---------------------------------------------------------------------------
__global__ void cvt_t_kernel(const float* __restrict__ src, bf16_t* __restrict__ dst,
                             int K, int N) {
  int id = blockIdx.x * blockDim.x + threadIdx.x;
  if (id >= K * N) return;
  int n = id / K, k = id % K;
  dst[id] = f2bf(src[(size_t)k * N + n]);
}

// ---------------------------------------------------------------------------
// cond = silu(time_embeds) @ cond_w + cond_b      (B x 2048)
// ---------------------------------------------------------------------------
__global__ void cond_mlp_kernel(const float* __restrict__ te,
                                const float* __restrict__ cw,
                                const float* __restrict__ cb,
                                float* __restrict__ cond) {
  int j = blockIdx.x * blockDim.x + threadIdx.x;
  if (j >= kB * 2 * kD) return;
  int b = j / (2 * kD), jj = j % (2 * kD);
  const float* tp = te + (size_t)b * kD;
  float acc = cb[jj];
  for (int i = 0; i < kD; ++i) {
    float v = tp[i];
    float s = v / (1.0f + __expf(-v));
    acc += s * cw[(size_t)i * 2 * kD + jj];
  }
  cond[j] = acc;
}

// ---------------------------------------------------------------------------
// LayerNorm + (1+scale)*x + bias -> bf16 units   (one block per token)
// ---------------------------------------------------------------------------
__global__ __launch_bounds__(256)
void ln_mod_kernel(const float* __restrict__ x,
                   const float* __restrict__ lnw,
                   const float* __restrict__ lnb,
                   const float* __restrict__ cond,
                   bf16_t* __restrict__ units) {
  __shared__ float s1[256], s2[256];
  int t = blockIdx.x;
  int b = t / kS;
  int tid = threadIdx.x;
  const float* xp = x + (size_t)t * kD;
  float4 xv = *(const float4*)(xp + tid * 4);
  s1[tid] = xv.x + xv.y + xv.z + xv.w;
  s2[tid] = xv.x * xv.x + xv.y * xv.y + xv.z * xv.z + xv.w * xv.w;
  __syncthreads();
  for (int st = 128; st > 0; st >>= 1) {
    if (tid < st) { s1[tid] += s1[tid + st]; s2[tid] += s2[tid + st]; }
    __syncthreads();
  }
  float mu  = s1[0] * (1.0f / kD);
  float var = s2[0] * (1.0f / kD) - mu * mu;
  float rin = rsqrtf(var + 1e-5f);
  const float* cd = cond + (size_t)b * 2 * kD;
  float xs[4] = {xv.x, xv.y, xv.z, xv.w};
#pragma unroll
  for (int e = 0; e < 4; ++e) {
    int c = tid * 4 + e;
    float u = (xs[e] - mu) * rin * lnw[c] + lnb[c];
    u = (1.0f + cd[c]) * u + cd[kD + c];
    units[(size_t)t * kD + c] = f2bf(u);
  }
}

// ---------------------------------------------------------------------------
// WMMA GEMM: C[M,N] = A[M,K] @ BT[N,K]^T, double-buffered TDM/async staging.
// 128 threads (4 waves); tile 64x64, BK=32.
// vT != nullptr: columns [2048,3072) scatter transposed to vT[b,h,d,s].
// ---------------------------------------------------------------------------
__global__ __launch_bounds__(128)
void gemm_bf16_wmma(const bf16_t* __restrict__ A, const bf16_t* __restrict__ BT,
                    float* __restrict__ Cf, bf16_t* __restrict__ Cb,
                    bf16_t* __restrict__ vT,
                    int M, int N, int K, int lda, int ldc) {
  __shared__ __align__(16) bf16_t As[2][64 * 32];
  __shared__ __align__(16) bf16_t Bs[2][64 * 32];
  const int tid  = threadIdx.x;
  const int wave = tid >> 5;
  const int lane = tid & 31;
  const int half = lane >> 4;
  const int nl   = lane & 15;
  const int m0 = blockIdx.y * 64;
  const int n0 = blockIdx.x * 64;

  const bf16_t* Ab = A  + (size_t)m0 * lda;
  const bf16_t* Bb = BT + (size_t)n0 * K;

  v8f acc[4] = {};

  // prologue: stage K-tile 0 into buffer 0
  stage_tile<64, 4>(As[0], Ab, lda, tid, wave);
  stage_tile<64, 4>(Bs[0], Bb, K,   tid, wave);

  const int iters = K / 32;
  for (int it = 0; it < iters; ++it) {
    const int buf = it & 1;
    const bool more = (it + 1) < iters;
    if (more) {
      stage_tile<64, 4>(As[buf ^ 1], Ab + (it + 1) * 32, lda, tid, wave);
      stage_tile<64, 4>(Bs[buf ^ 1], Bb + (it + 1) * 32, K,   tid, wave);
      __builtin_prefetch(Ab + (size_t)(tid >> 1) * lda + (it + 1) * 32, 0, 1);
    }
    stage_wait<2, 4>(more, wave);   // current buffer landed

    V16BF fa;
    const int arow = wave * 16 + nl;
    fa.u[0] = *(const uint4*)&As[buf][arow * 32 + half * 8];
    fa.u[1] = *(const uint4*)&As[buf][arow * 32 + 16 + half * 8];
#pragma unroll
    for (int nf = 0; nf < 4; ++nf) {
      V16BF fb;
      const bf16_t* bp = &Bs[buf][(nf * 16 + nl) * 32 + half * 8];
      fb.u[0] = *(const uint4*)bp;
      fb.u[1] = *(const uint4*)(bp + 16);
      acc[nf] = __builtin_amdgcn_wmma_f32_16x16x32_bf16(
          false, fa.v, false, fb.v, (short)0, acc[nf], false, false);
    }
    __syncthreads();                // all waves done reading buf before restage
  }

  const bool isv = (vT != nullptr) && (n0 >= 2 * kD) && (n0 < 3 * kD);
#pragma unroll
  for (int nf = 0; nf < 4; ++nf) {
#pragma unroll
    for (int v = 0; v < 8; ++v) {
      int gr = m0 + wave * 16 + v + half * 8;
      int gc = n0 + nf * 16 + nl;
      float val = acc[nf][v];
      if (isv) {
        int hh = (gc - 2 * kD) >> 6, dd = (gc - 2 * kD) & 63;
        int bb = gr >> 11, ss = gr & (kS - 1);
        vT[(size_t)((bb * kH + hh) * kHD + dd) * kS + ss] = f2bf(val);
      } else if (Cb) {
        Cb[(size_t)gr * ldc + gc] = f2bf(val);
      } else {
        Cf[(size_t)gr * ldc + gc] = val;
      }
    }
  }
}

// ---------------------------------------------------------------------------
// RoPE on q,k (first 32 dims of each head), in place on bf16 proj buffer
// ---------------------------------------------------------------------------
__global__ void rope_kernel(bf16_t* __restrict__ proj) {
  int id = blockIdx.x * blockDim.x + threadIdx.x;
  if (id >= kM * kH * 16) return;
  int i = id & 15;
  int h = (id >> 4) & 15;
  int t = id >> 8;
  int pos = t % kS;
  float invf = __powf(10000.0f, -(float)i / 16.0f);
  float a = (float)pos * invf;
  float cs = __cosf(a), sn = __sinf(a);
  bf16_t* qb = proj + (size_t)t * kNSUM + h * kHD;
  bf16_t* kb = qb + kD;
  float xl = bf2f(qb[i]), xr = bf2f(qb[16 + i]);
  qb[i]      = f2bf(xl * cs - xr * sn);
  qb[16 + i] = f2bf(xr * cs + xl * sn);
  xl = bf2f(kb[i]); xr = bf2f(kb[16 + i]);
  kb[i]      = f2bf(xl * cs - xr * sn);
  kb[16 + i] = f2bf(xr * cs + xl * sn);
}

// ---------------------------------------------------------------------------
// Flash attention, double-buffered K/V tiles via TDM/async.
// grid (S/64, H, B), 128 threads (4 waves x 16 q rows)
// ---------------------------------------------------------------------------
__global__ __launch_bounds__(128)
void attn_kernel(const bf16_t* __restrict__ proj, const bf16_t* __restrict__ vT,
                 bf16_t* __restrict__ concat) {
  __shared__ __align__(16) bf16_t Qs[64 * 64];
  __shared__ __align__(16) bf16_t Ks[2][64 * 64];
  __shared__ __align__(16) bf16_t Vs[2][64 * 64];   // [d][key]
  __shared__ __align__(16) bf16_t Ps[4 * 16 * 64];

  const int tid  = threadIdx.x;
  const int wave = tid >> 5, lane = tid & 31, half = lane >> 4, nl = lane & 15;
  const int qt = blockIdx.x;
  const int h  = blockIdx.y;
  const int b  = blockIdx.z;
  const int q0 = qt * 64;
  const size_t rs = kNSUM;
  const bf16_t* qbase = proj + ((size_t)b * kS) * rs + (size_t)h * kHD;
  const bf16_t* kbase = qbase + kD;
  const bf16_t* vTb   = vT + (size_t)(b * kH + h) * kHD * kS;

  // stage Q tile + first K/V tiles
  stage_tile<64, 8>(Qs, qbase + (size_t)q0 * rs, (int)rs, tid, wave);
  stage_tile<64, 8>(Ks[0], kbase, (int)rs, tid, wave);
  stage_tile<64, 8>(Vs[0], vTb, kS, tid, wave);

  float run_max[8], run_sum[8];
#pragma unroll
  for (int v = 0; v < 8; ++v) { run_max[v] = -1e30f; run_sum[v] = 0.0f; }
  v8f o[4] = {};
  const float scale = 0.125f;   // HEAD_DIM^-0.5
  V16BF fq[2];
  bool have_q = false;

  const int iters = kS / 64;
  for (int it = 0; it < iters; ++it) {
    const int buf = it & 1;
    const bool more = (it + 1) < iters;
    if (more) {
      stage_tile<64, 8>(Ks[buf ^ 1], kbase + (size_t)(it + 1) * 64 * rs, (int)rs, tid, wave);
      stage_tile<64, 8>(Vs[buf ^ 1], vTb + (it + 1) * 64, kS, tid, wave);
    }
    stage_wait<2, 8>(more, wave);

    if (!have_q) {   // Q landed with the first wait
      have_q = true;
      int arow = wave * 16 + nl;
#pragma unroll
      for (int c = 0; c < 2; ++c) {
        fq[c].u[0] = *(const uint4*)&Qs[arow * 64 + c * 32 + half * 8];
        fq[c].u[1] = *(const uint4*)&Qs[arow * 64 + c * 32 + 16 + half * 8];
      }
    }

    // scores = Q @ K^T
    v8f sc[4] = {};
#pragma unroll
    for (int nf = 0; nf < 4; ++nf) {
      const bf16_t* kp = &Ks[buf][(nf * 16 + nl) * 64];
#pragma unroll
      for (int c = 0; c < 2; ++c) {
        V16BF fk;
        fk.u[0] = *(const uint4*)(kp + c * 32 + half * 8);
        fk.u[1] = *(const uint4*)(kp + c * 32 + 16 + half * 8);
        sc[nf] = __builtin_amdgcn_wmma_f32_16x16x32_bf16(
            false, fq[c].v, false, fk.v, (short)0, sc[nf], false, false);
      }
    }

    // online softmax
    float corr[8];
#pragma unroll
    for (int v = 0; v < 8; ++v) {
      float mx = -1e30f;
#pragma unroll
      for (int nf = 0; nf < 4; ++nf) mx = fmaxf(mx, sc[nf][v] * scale);
#pragma unroll
      for (int off = 1; off < 16; off <<= 1) mx = fmaxf(mx, __shfl_xor(mx, off, 32));
      float nm = fmaxf(run_max[v], mx);
      corr[v] = __expf(run_max[v] - nm);
      run_max[v] = nm;
    }
    float rsum[8];
#pragma unroll
    for (int v = 0; v < 8; ++v) rsum[v] = 0.0f;
#pragma unroll
    for (int nf = 0; nf < 4; ++nf) {
#pragma unroll
      for (int v = 0; v < 8; ++v) {
        float p = __expf(sc[nf][v] * scale - run_max[v]);
        rsum[v] += p;
        int m = v + half * 8;
        Ps[wave * 1024 + m * 64 + nf * 16 + nl] = f2bf(p);
      }
    }
#pragma unroll
    for (int v = 0; v < 8; ++v) {
      float s = rsum[v];
#pragma unroll
      for (int off = 1; off < 16; off <<= 1) s += __shfl_xor(s, off, 32);
      run_sum[v] = run_sum[v] * corr[v] + s;
#pragma unroll
      for (int nf = 0; nf < 4; ++nf) o[nf][v] *= corr[v];
    }

    // O += P @ V
    const bf16_t* pw = &Ps[wave * 1024];
    V16BF fp[2];
#pragma unroll
    for (int c = 0; c < 2; ++c) {
      fp[c].u[0] = *(const uint4*)(pw + nl * 64 + c * 32 + half * 8);
      fp[c].u[1] = *(const uint4*)(pw + nl * 64 + c * 32 + 16 + half * 8);
    }
#pragma unroll
    for (int nf = 0; nf < 4; ++nf) {
      const bf16_t* vp = &Vs[buf][(nf * 16 + nl) * 64];
#pragma unroll
      for (int c = 0; c < 2; ++c) {
        V16BF fv;
        fv.u[0] = *(const uint4*)(vp + c * 32 + half * 8);
        fv.u[1] = *(const uint4*)(vp + c * 32 + 16 + half * 8);
        o[nf] = __builtin_amdgcn_wmma_f32_16x16x32_bf16(
            false, fp[c].v, false, fv.v, (short)0, o[nf], false, false);
      }
    }
    __syncthreads();   // done reading buf before restage
  }

#pragma unroll
  for (int v = 0; v < 8; ++v) {
    float inv = 1.0f / run_sum[v];
    int m  = v + half * 8;
    int gr = b * kS + q0 + wave * 16 + m;
#pragma unroll
    for (int nf = 0; nf < 4; ++nf) {
      concat[(size_t)gr * kD + h * kHD + nf * 16 + nl] = f2bf(o[nf][v] * inv);
    }
  }
}

// ---------------------------------------------------------------------------
// ffh = ff * gelu_tanh(ff_gate)
// ---------------------------------------------------------------------------
__global__ void ffh_kernel(const bf16_t* __restrict__ proj, bf16_t* __restrict__ ffh) {
  int id = blockIdx.x * blockDim.x + threadIdx.x;
  if (id >= kM * kFF) return;
  int t = id / kFF, j = id % kFF;
  const bf16_t* row = proj + (size_t)t * kNSUM;
  float f = bf2f(row[3 * kD + j]);
  float g = bf2f(row[3 * kD + kFF + j]);
  float gl = 0.5f * g * (1.0f + tanhf(0.7978845608f * (g + 0.044715f * g * g * g)));
  ffh[(size_t)t * kFF + j] = f2bf(f * gl);
}

// ---------------------------------------------------------------------------
// out = inputs + attn_out + ff_out + b_ff
// ---------------------------------------------------------------------------
__global__ void final_add_kernel(const float* __restrict__ x,
                                 const float* __restrict__ ao,
                                 const float* __restrict__ fo,
                                 const float* __restrict__ bff,
                                 float* __restrict__ out) {
  int id = blockIdx.x * blockDim.x + threadIdx.x;
  if (id >= kM * kD) return;
  int c = id % kD;
  out[id] = x[id] + ao[id] + fo[id] + bff[c];
}

// ---------------------------------------------------------------------------
extern "C" void kernel_launch(void* const* d_in, const int* in_sizes, int n_in,
                              void* d_out, int out_size, void* d_ws, size_t ws_size,
                              hipStream_t stream) {
  (void)in_sizes; (void)n_in; (void)out_size; (void)ws_size;
  const float* inputs = (const float*)d_in[0];
  const float* te     = (const float*)d_in[1];
  const float* ln_w   = (const float*)d_in[2];
  const float* ln_b   = (const float*)d_in[3];
  const float* cond_w = (const float*)d_in[4];
  const float* cond_b = (const float*)d_in[5];
  const float* w_in   = (const float*)d_in[6];
  const float* w_attn = (const float*)d_in[7];
  const float* w_ff   = (const float*)d_in[8];
  const float* b_ff   = (const float*)d_in[9];
  float* out = (float*)d_out;

  char* ws = (char*)d_ws;
  size_t off = 0;
  auto alloc = [&](size_t bytes) -> char* {
    char* p = ws + off;
    off += (bytes + 255) & ~(size_t)255;
    return p;
  };
  float*  cond     = (float*)alloc((size_t)kB * 2 * kD * sizeof(float));
  bf16_t* units    = (bf16_t*)alloc((size_t)kM * kD * 2);
  bf16_t* w_inT    = (bf16_t*)alloc((size_t)kD * kNSUM * 2);   // [N][K]
  bf16_t* w_attnT  = (bf16_t*)alloc((size_t)kD * kD * 2);
  bf16_t* w_ffT    = (bf16_t*)alloc((size_t)kFF * kD * 2);
  bf16_t* proj     = (bf16_t*)alloc((size_t)kM * kNSUM * 2);
  bf16_t* vT       = (bf16_t*)alloc((size_t)kB * kH * kHD * kS * 2);
  bf16_t* concat   = (bf16_t*)alloc((size_t)kM * kD * 2);
  bf16_t* ffh      = (bf16_t*)alloc((size_t)kM * kFF * 2);
  float*  attn_out = (float*)alloc((size_t)kM * kD * 4);
  float*  ff_out   = (float*)alloc((size_t)kM * kD * 4);

  // 1. weight conversion fp32 -> bf16, transposed to [N][K]
  cvt_t_kernel<<<(kD * kNSUM + 255) / 256, 256, 0, stream>>>(w_in, w_inT, kD, kNSUM);
  cvt_t_kernel<<<(kD * kD + 255) / 256, 256, 0, stream>>>(w_attn, w_attnT, kD, kD);
  cvt_t_kernel<<<(kFF * kD + 255) / 256, 256, 0, stream>>>(w_ff, w_ffT, kFF, kD);

  // 2. conditioning MLP, 3. LN + modulation
  cond_mlp_kernel<<<(kB * 2 * kD + 255) / 256, 256, 0, stream>>>(te, cond_w, cond_b, cond);
  ln_mod_kernel<<<kM, 256, 0, stream>>>(inputs, ln_w, ln_b, cond, units);

  // 4. fused projection GEMM (4096 x 11264 x 1024); V columns scatter to vT
  gemm_bf16_wmma<<<dim3(kNSUM / 64, kM / 64), 128, 0, stream>>>(
      units, w_inT, nullptr, proj, vT, kM, kNSUM, kD, kD, kNSUM);

  // 5. RoPE on q,k
  rope_kernel<<<(kM * kH * 16 + 255) / 256, 256, 0, stream>>>(proj);

  // 6. flash attention
  attn_kernel<<<dim3(kS / 64, kH, kB), 128, 0, stream>>>(proj, vT, concat);

  // 7. GEGLU elementwise
  ffh_kernel<<<(kM * kFF + 255) / 256, 256, 0, stream>>>(proj, ffh);

  // 8. attn_out GEMM (4096 x 1024 x 1024)
  gemm_bf16_wmma<<<dim3(kD / 64, kM / 64), 128, 0, stream>>>(
      concat, w_attnT, attn_out, nullptr, nullptr, kM, kD, kD, kD, kD);

  // 9. FFN-out GEMM (4096 x 1024 x 4096)
  gemm_bf16_wmma<<<dim3(kD / 64, kM / 64), 128, 0, stream>>>(
      ffh, w_ffT, ff_out, nullptr, nullptr, kM, kD, kFF, kFF, kD);

  // 10. residual add
  final_add_kernel<<<(kM * kD + 255) / 256, 256, 0, stream>>>(
      inputs, attn_out, ff_out, b_ff, out);
}
